// RelGraphConvLayer_31688268709949
// MI455X (gfx1250) — matrix operand order repaired and checked by reading, not compiled, not run on hardware
//
#include <hip/hip_runtime.h>
#include <stdint.h>

#define D_OUT        256
#define K_DIM        512
#define NCHUNK_COLS  128
#define LDS_STRIDE   520   // 512 + 8 bf16 -> 1040B row, 260 dwords (≡4 mod 64 banks), 16B aligned

typedef __attribute__((ext_vector_type(16))) __bf16 v16bf;
typedef __attribute__((ext_vector_type(8)))  float  v8f;

// round-to-nearest-even f32 -> bf16, result kept in the TOP 16 bits of a dword
__device__ __forceinline__ uint32_t bf16_rne_hi(float f) {
  uint32_t u = __float_as_uint(f);
  return (u + 0x7FFFu + ((u >> 16) & 1u)) & 0xFFFF0000u;
}

__global__ void init_out_kernel(float* __restrict__ out, const float* __restrict__ bias, int total) {
  int i = blockIdx.x * blockDim.x + threadIdx.x;
  if (i < total) out[i] = bias[i & (D_OUT - 1)];
}

__global__ void zero_kernel(float* __restrict__ p, int n) {
  int i = blockIdx.x * blockDim.x + threadIdx.x;
  if (i < n) p[i] = 0.0f;
}

__global__ void deg_kernel(const int* __restrict__ dst_idx, float* __restrict__ deg, int E) {
  int i = blockIdx.x * blockDim.x + threadIdx.x;
  if (i < E) unsafeAtomicAdd(&deg[dst_idx[i]], 1.0f);
}

__global__ void norm_kernel(float* __restrict__ deg, int n) {
  int i = blockIdx.x * blockDim.x + threadIdx.x;
  if (i < n) {
    float d = deg[i];
    deg[i] = (d > 0.0f) ? rsqrtf(d) : 0.0f;
  }
}

__global__ __launch_bounds__(256)
void gconv_wmma_kernel(const float* __restrict__ src_feats,
                       const float* __restrict__ edge_feats,
                       const int*   __restrict__ src_idx,
                       const int*   __restrict__ dst_idx,
                       const float* __restrict__ weights,
                       const float* __restrict__ norm,
                       float*       __restrict__ out,
                       int E, int numBlocks)
{
  __shared__ uint16_t ldsHi[NCHUNK_COLS * LDS_STRIDE];  // W chunk, bf16 high part, [n][k]
  __shared__ uint16_t ldsLo[NCHUNK_COLS * LDS_STRIDE];  // W chunk, bf16 residual,  [n][k]

  const int tid   = threadIdx.x;
  const int nbase = blockIdx.x * NCHUNK_COLS;

  // ---- Stage W chunk into LDS (one time; reused across all edge blocks) ----
  for (int i = tid; i < K_DIM * NCHUNK_COLS; i += 256) {
    int k = i >> 7;                  // consecutive tids -> consecutive n (coalesced global read)
    int n = i & (NCHUNK_COLS - 1);
    float f = weights[k * D_OUT + nbase + n];
    uint32_t hb = bf16_rne_hi(f);
    float    lo = f - __uint_as_float(hb);
    uint32_t lb = bf16_rne_hi(lo);
    ldsHi[n * LDS_STRIDE + k] = (uint16_t)(hb >> 16);
    ldsLo[n * LDS_STRIDE + k] = (uint16_t)(lb >> 16);
  }
  __syncthreads();

  const int wave  = tid >> 5;
  const int lane  = tid & 31;
  const int nl    = lane & 15;   // A: row M; B/C: column within 16-tile
  const int khalf = lane >> 4;   // which K-half of the fragment this lane holds

  for (int eb = blockIdx.y; eb < numBlocks; eb += gridDim.y) {
    const int e0 = eb * 128 + wave * 16;

    // A-gather row pointers for this lane's row (M = nl)
    int eA  = e0 + nl;
    int eAc = (eA < E) ? eA : (E - 1);
    const float* prow_src  = src_feats  + (size_t)src_idx[eAc] * 256;
    const float* prow_edge = edge_feats + (size_t)eAc * 256;

    // dst index + norm for the 8 C-rows (edges) this lane accumulates
    int dd[8]; float nm[8];
    #pragma unroll
    for (int r = 0; r < 8; ++r) {
      int e = e0 + khalf * 8 + r;
      if (e < E) { int d = dst_idx[e]; dd[r] = d; nm[r] = norm[d]; }
      else       { dd[r] = 0; nm[r] = 0.0f; }
    }

    v8f acc[8];
    #pragma unroll
    for (int nt = 0; nt < 8; ++nt) acc[nt] = (v8f)(0.0f);

    for (int kt = 0; kt < 16; ++kt) {          // K = 512 in 16x32 steps
      const int kb = kt * 32;                  // uniform across wave
      const float* rowp = (kb < 256) ? (prow_src  + kb         + khalf * 8)
                                     : (prow_edge + (kb - 256) + khalf * 8);
      // two 8-float runs per lane: K = kb + khalf*8 + [0,8) and kb + 16 + khalf*8 + [0,8)
      float4 f0 = *(const float4*)(rowp);
      float4 f1 = *(const float4*)(rowp + 4);
      float4 f2 = *(const float4*)(rowp + 16);
      float4 f3 = *(const float4*)(rowp + 20);
      float x[16] = { f0.x, f0.y, f0.z, f0.w,  f1.x, f1.y, f1.z, f1.w,
                      f2.x, f2.y, f2.z, f2.w,  f3.x, f3.y, f3.z, f3.w };

      union { uint32_t d[8]; v16bf v; } ah, al;
      #pragma unroll
      for (int j = 0; j < 8; ++j) {
        uint32_t h0 = bf16_rne_hi(x[2 * j]);
        uint32_t h1 = bf16_rne_hi(x[2 * j + 1]);
        ah.d[j] = (h0 >> 16) | h1;
        uint32_t l0 = bf16_rne_hi(x[2 * j]     - __uint_as_float(h0));
        uint32_t l1 = bf16_rne_hi(x[2 * j + 1] - __uint_as_float(h1));
        al.d[j] = (l0 >> 16) | l1;
      }

      const int koff = kt * 32 + khalf * 8;
      #pragma unroll
      for (int nt = 0; nt < 8; ++nt) {
        const uint16_t* bp = &ldsHi[(nt * 16 + nl) * LDS_STRIDE + koff];
        const uint16_t* bq = &ldsLo[(nt * 16 + nl) * LDS_STRIDE + koff];
        union { uint4 q[2]; v16bf v; } bh, bl;
        bh.q[0] = *(const uint4*)(bp);
        bh.q[1] = *(const uint4*)(bp + 16);
        bl.q[0] = *(const uint4*)(bq);
        bl.q[1] = *(const uint4*)(bq + 16);
        // hi*hi + hi*lo + lo*hi  (bf16x3 split ~ f32 accuracy)
        acc[nt] = __builtin_amdgcn_wmma_f32_16x16x32_bf16(false, ah.v, false, bh.v,
                                                          (short)0, acc[nt], false, false);
        acc[nt] = __builtin_amdgcn_wmma_f32_16x16x32_bf16(false, ah.v, false, bl.v,
                                                          (short)0, acc[nt], false, false);
        acc[nt] = __builtin_amdgcn_wmma_f32_16x16x32_bf16(false, al.v, false, bh.v,
                                                          (short)0, acc[nt], false, false);
      }
    }

    // ---- normalized atomic scatter into h[dst] ----
    #pragma unroll
    for (int nt = 0; nt < 8; ++nt) {
      int col = nbase + nt * 16 + nl;
      #pragma unroll
      for (int r = 0; r < 8; ++r) {
        float v = acc[nt][r] * nm[r];
        unsafeAtomicAdd(out + (size_t)dd[r] * D_OUT + col, v);
      }
    }
  }
}

extern "C" void kernel_launch(void* const* d_in, const int* in_sizes, int n_in,
                              void* d_out, int out_size, void* d_ws, size_t ws_size,
                              hipStream_t stream) {
  const float* src_feats  = (const float*)d_in[0];
  const float* edge_feats = (const float*)d_in[1];
  const int*   src_idx    = (const int*)d_in[2];
  const int*   dst_idx    = (const int*)d_in[3];
  const float* weights    = (const float*)d_in[4];
  const float* h_bias     = (const float*)d_in[5];
  // d_in[6] = num_dst (device scalar) — derive from out_size instead (host-visible)
  const int E     = in_sizes[2];
  const int n_dst = out_size / D_OUT;

  float* out = (float*)d_out;
  float* deg = (float*)d_ws;   // n_dst floats: degree, then norm in place

  const int tot = n_dst * D_OUT;
  init_out_kernel<<<(tot + 255) / 256, 256, 0, stream>>>(out, h_bias, tot);
  zero_kernel<<<(n_dst + 255) / 256, 256, 0, stream>>>(deg, n_dst);
  deg_kernel<<<(E + 255) / 256, 256, 0, stream>>>(dst_idx, deg, E);
  norm_kernel<<<(n_dst + 255) / 256, 256, 0, stream>>>(deg, n_dst);

  const int numBlocks = (E + 127) / 128;       // 128 edges per workgroup iteration
  dim3 grid(2, 160);                            // 2 N-chunks x 160 persistent blocks
  gconv_wmma_kernel<<<grid, 256, 0, stream>>>(src_feats, edge_feats, src_idx, dst_idx,
                                              weights, deg, out, E, numBlocks);
}
